// GNN_22007412425053
// MI455X (gfx1250) — matrix operand order repaired and checked by reading, not compiled, not run on hardware
//
#include <hip/hip_runtime.h>
#include <hip/hip_bf16.h>

// ---------------------------------------------------------------------------
// GNN message passing (D=128) on gfx1250: all GEMMs via v_wmma_f32_16x16x32_f16
// ---------------------------------------------------------------------------

#define D      128
#define F      10
#define N_IP   20000
#define N_CONN 100000
#define NEDGE  200000
#define T_ITER 3

typedef __attribute__((ext_vector_type(16))) _Float16 v16h;
typedef __attribute__((ext_vector_type(8)))  _Float16 v8h;
typedef __attribute__((ext_vector_type(8)))  float    v8f;

// ---- WMMA helpers ---------------------------------------------------------

__device__ inline v8f wmma16(v16h a, v16h b, v8f c) {
  // D = A(16x32 f16) * B(32x16 f16) + C(16x16 f32)
  return __builtin_amdgcn_wmma_f32_16x16x32_f16(
      /*neg_a=*/false, a, /*neg_b=*/false, b,
      /*c_mod=*/(short)0, c, /*reuse_a=*/false, /*reuse_b=*/false);
}

// A fragment: 16 rows (m=0..15) x K-window [k0,k0+32) from row-major f16 buf.
// Lanes 0-15 hold K {k0..k0+15}; lanes 16-31 hold K {k0+16..k0+31}.
__device__ inline v16h frag_a(const _Float16* buf, int stride, int k0, int lane) {
  int m  = lane & 15;
  int kh = (lane >> 4) << 4;                 // 0 or 16
  const _Float16* p = buf + m * stride + k0 + kh;
  v8h lo = *(const v8h*)(p);
  v8h hi = *(const v8h*)(p + 8);
  v16h r;
#pragma unroll
  for (int i = 0; i < 8; ++i) { r[i] = lo[i]; r[i + 8] = hi[i]; }
  return r;
}

// B fragment: cols n0..n0+15, K-window [k0,k0+32) from Wt stored [N][K] f16.
// Lanes 0-15 hold K {k0..k0+15} for col n; lanes 16-31 hold K {k0+16..k0+31}.
__device__ inline v16h frag_b(const _Float16* wt, int K, int n0, int k0, int lane) {
  int n  = lane & 15;
  int kh = (lane >> 4) << 4;                 // 0 or 16
  return *(const v16h*)(wt + (size_t)(n0 + n) * K + k0 + kh);
}

// ---- setup kernels --------------------------------------------------------

// dst[n*K + k] = (n < N) ? src[k*N + n] : 0   (f32 -> f16, transpose, pad N)
__global__ void k_transpose_f16(const float* __restrict__ src,
                                _Float16* __restrict__ dst,
                                int K, int N, int Npad) {
  int i = blockIdx.x * blockDim.x + threadIdx.x;
  if (i >= Npad * K) return;
  int n = i / K, k = i - n * K;
  dst[(size_t)n * K + k] = (_Float16)((n < N) ? src[(size_t)k * N + n] : 0.0f);
}

__global__ void k_init_states(const float* __restrict__ feat,
                              float* __restrict__ ip_state,
                              float* __restrict__ conn_state) {
  int i = blockIdx.x * blockDim.x + threadIdx.x;
  if (i < N_IP * D) ip_state[i] = 1.0f;
  if (i < N_CONN * D) {
    int node = i >> 7, c = i & (D - 1);
    conn_state[i] = (c < F) ? feat[node * F + c] : 0.0f;
  }
}

__global__ void k_count(const int* __restrict__ idx, float* __restrict__ cnt, int n) {
  int i = blockIdx.x * blockDim.x + threadIdx.x;
  if (i < n) unsafeAtomicAdd(&cnt[idx[i]], 1.0f);
}

// ---- edge message kernel --------------------------------------------------
// 64 edges/block (4 row tiles). A = [ stateA[idxA[e]] | stateB[idxB[e]] ]
// (64 x 256 f16 in LDS). Each wave owns one 16-col output tile; every B
// fragment is loaded once and reused across the 4 row-tile accumulators.
// out = relu(A @ Wt^T + bias), scatter-added into sum_out[idxB[e]].
__global__ __launch_bounds__(256) void k_msg(
    const float* __restrict__ stateA, const int* __restrict__ idxA,
    const float* __restrict__ stateB, const int* __restrict__ idxB,
    const _Float16* __restrict__ wt,      // [128][256]
    const float* __restrict__ bias,       // [128]
    float* __restrict__ sum_out,          // [n_seg][128]
    int n_edges) {
  __shared__ _Float16 A[64 * 256];        // 32 KB

  int tile = blockIdx.x << 6;
  int tid  = threadIdx.x;

  // stage gathered activations: 4 threads per edge row, 64 cols each
  {
    int row = tid >> 2;                   // 0..63
    int c0  = (tid & 3) << 6;             // 0,64,128,192
    int e   = tile + row;
    bool ok = e < n_edges;
    const float* srcp;
    if (c0 < D) {
      int ia = ok ? idxA[e] : 0;
      srcp = stateA + (size_t)ia * D + c0;
    } else {
      int ib = ok ? idxB[e] : 0;
      srcp = stateB + (size_t)ib * D + (c0 - D);
    }
    _Float16* dstp = A + row * 256 + c0;
#pragma unroll
    for (int i = 0; i < 64; i += 4) {
      float4 v = ok ? *(const float4*)(srcp + i) : float4{0.f, 0.f, 0.f, 0.f};
      dstp[i + 0] = (_Float16)v.x;
      dstp[i + 1] = (_Float16)v.y;
      dstp[i + 2] = (_Float16)v.z;
      dstp[i + 3] = (_Float16)v.w;
    }
  }
  __syncthreads();

  int wave = tid >> 5, lane = tid & 31;
  int n0 = wave << 4;                     // each wave owns one 16-col tile
  v8f acc0 = {}, acc1 = {}, acc2 = {}, acc3 = {};
#pragma unroll
  for (int ks = 0; ks < 8; ++ks) {        // K = 256
    v16h b = frag_b(wt, 256, n0, ks * 32, lane);   // fetched once, used 4x
    acc0 = wmma16(frag_a(A +  0 * 256, 256, ks * 32, lane), b, acc0);
    acc1 = wmma16(frag_a(A + 16 * 256, 256, ks * 32, lane), b, acc1);
    acc2 = wmma16(frag_a(A + 32 * 256, 256, ks * 32, lane), b, acc2);
    acc3 = wmma16(frag_a(A + 48 * 256, 256, ks * 32, lane), b, acc3);
  }

  int n   = n0 + (lane & 15);
  float bv = bias[n];
  auto emit = [&](v8f acc, int rt) {
#pragma unroll
    for (int r = 0; r < 8; ++r) {
      int m = r + ((lane >> 4) << 3);
      int e = tile + (rt << 4) + m;
      if (e < n_edges) {
        float v = acc[r] + bv;
        v = v > 0.0f ? v : 0.0f;
        unsafeAtomicAdd(&sum_out[(size_t)idxB[e] * D + n], v);
      }
    }
  };
  emit(acc0, 0); emit(acc1, 1); emit(acc2, 2); emit(acc3, 3);
}

// ---- GRU update kernel ----------------------------------------------------
// x = sum/max(cnt,1); xm = x@K + b0; hm = h@R + b1 (reset_after=True)
// z = sig(xm_z+hm_z); r = sig(xm_r+hm_r); hc = tanh(xm_h + r*hm_h)
// h' = z*h + (1-z)*hc
__global__ __launch_bounds__(256) void k_gru(
    const float* __restrict__ sum, const float* __restrict__ cnt,
    float* __restrict__ h_state,
    const _Float16* __restrict__ kt,      // [384][128] transposed kernel
    const _Float16* __restrict__ rt,      // [384][128] transposed rec kernel
    const float* __restrict__ bias,       // [2][384]
    int n_nodes) {
  __shared__ _Float16 X[16 * 128];
  __shared__ _Float16 H[16 * 128];
  __shared__ float    ZR[16 * 256];       // fused xm+hm for z,r gates
  __shared__ float    XH[16 * 128];       // xm, candidate part
  __shared__ float    HH[16 * 128];       // hm, candidate part

  int tile = blockIdx.x << 4;
  int tid  = threadIdx.x;

#pragma unroll
  for (int i = 0; i < 8; ++i) {
    int p = (tid << 3) + i;                // 0..2047
    int m = p >> 7, c = p & (D - 1);
    int node = tile + m;
    float xv = 0.0f, hv = 0.0f;
    if (node < n_nodes) {
      float cn = cnt[node]; cn = cn > 1.0f ? cn : 1.0f;
      xv = sum[(size_t)node * D + c] / cn;
      hv = h_state[(size_t)node * D + c];
    }
    X[m * D + c] = (_Float16)xv;
    H[m * D + c] = (_Float16)hv;
  }
  __syncthreads();

  int wave = tid >> 5, lane = tid & 31;
  // 32 wave-jobs: 0..15 ZR tiles (cols 0..255), 16..23 XH, 24..31 HH
#pragma unroll
  for (int j = 0; j < 4; ++j) {
    int job = (wave << 2) + j;             // uniform per wave -> EXEC stays full
    v8f acc = {};
    if (job < 16) {
      int n0 = job << 4;
#pragma unroll
      for (int ks = 0; ks < 4; ++ks)
        acc = wmma16(frag_a(X, D, ks * 32, lane), frag_b(kt, D, n0, ks * 32, lane), acc);
#pragma unroll
      for (int ks = 0; ks < 4; ++ks)
        acc = wmma16(frag_a(H, D, ks * 32, lane), frag_b(rt, D, n0, ks * 32, lane), acc);
      int n = lane & 15;
#pragma unroll
      for (int r = 0; r < 8; ++r) {
        int m = r + ((lane >> 4) << 3);
        ZR[m * 256 + n0 + n] = acc[r];
      }
    } else if (job < 24) {
      int n0 = 256 + ((job - 16) << 4);
#pragma unroll
      for (int ks = 0; ks < 4; ++ks)
        acc = wmma16(frag_a(X, D, ks * 32, lane), frag_b(kt, D, n0, ks * 32, lane), acc);
      int n = lane & 15;
#pragma unroll
      for (int r = 0; r < 8; ++r) {
        int m = r + ((lane >> 4) << 3);
        XH[m * D + (n0 - 256) + n] = acc[r];
      }
    } else {
      int n0 = 256 + ((job - 24) << 4);
#pragma unroll
      for (int ks = 0; ks < 4; ++ks)
        acc = wmma16(frag_a(H, D, ks * 32, lane), frag_b(rt, D, n0, ks * 32, lane), acc);
      int n = lane & 15;
#pragma unroll
      for (int r = 0; r < 8; ++r) {
        int m = r + ((lane >> 4) << 3);
        HH[m * D + (n0 - 256) + n] = acc[r];
      }
    }
  }
  __syncthreads();

#pragma unroll
  for (int i = 0; i < 8; ++i) {
    int p = (tid << 3) + i;
    int m = p >> 7, c = p & (D - 1);
    int node = tile + m;
    if (node < n_nodes) {
      float z = ZR[m * 256 + c] + bias[c] + bias[384 + c];
      z = 1.0f / (1.0f + __expf(-z));
      float rg = ZR[m * 256 + 128 + c] + bias[128 + c] + bias[384 + 128 + c];
      rg = 1.0f / (1.0f + __expf(-rg));
      float hc = tanhf(XH[m * D + c] + bias[256 + c] +
                       rg * (HH[m * D + c] + bias[384 + 256 + c]));
      float h = h_state[(size_t)node * D + c];
      h_state[(size_t)node * D + c] = z * h + (1.0f - z) * hc;
    }
  }
}

// ---- readout: relu(s@W1)->relu(@W2)->@W3->softmax -------------------------
// 32 connections/block (2 row tiles); B fragments reused across both tiles.
__global__ __launch_bounds__(256) void k_readout(
    const float* __restrict__ state,
    const _Float16* __restrict__ w1t, const float* __restrict__ b1,  // [128][128]
    const _Float16* __restrict__ w2t, const float* __restrict__ b2,  // [64][128]
    const _Float16* __restrict__ w3t, const float* __restrict__ b3,  // [16][64]
    float* __restrict__ out, int n_nodes) {
  __shared__ _Float16 S[32 * 128];
  __shared__ _Float16 H1[32 * 128];
  __shared__ _Float16 H2[32 * 64];
  __shared__ float    LG[32 * 16];

  int tile = blockIdx.x << 5;
  int tid  = threadIdx.x;

#pragma unroll
  for (int i = 0; i < 16; ++i) {
    int p = (tid << 4) + i;
    int m = p >> 7, c = p & (D - 1);
    int node = tile + m;
    S[m * D + c] = (_Float16)((node < n_nodes) ? state[(size_t)node * D + c] : 0.0f);
  }
  __syncthreads();

  int wave = tid >> 5, lane = tid & 31;

  { // layer 1: 32x128, all 8 waves, 2 row tiles per wave
    int n0 = wave << 4;
    v8f a0 = {}, a1 = {};
#pragma unroll
    for (int ks = 0; ks < 4; ++ks) {
      v16h b = frag_b(w1t, D, n0, ks * 32, lane);
      a0 = wmma16(frag_a(S,            D, ks * 32, lane), b, a0);
      a1 = wmma16(frag_a(S + 16 * D,   D, ks * 32, lane), b, a1);
    }
    int n = n0 + (lane & 15);
    float b = b1[n];
#pragma unroll
    for (int r = 0; r < 8; ++r) {
      int m = r + ((lane >> 4) << 3);
      float v0 = a0[r] + b, v1 = a1[r] + b;
      H1[m * D + n]        = (_Float16)(v0 > 0.0f ? v0 : 0.0f);
      H1[(m + 16) * D + n] = (_Float16)(v1 > 0.0f ? v1 : 0.0f);
    }
  }
  __syncthreads();

  if (wave < 4) { // layer 2: 32x64
    int n0 = wave << 4;
    v8f a0 = {}, a1 = {};
#pragma unroll
    for (int ks = 0; ks < 4; ++ks) {
      v16h b = frag_b(w2t, D, n0, ks * 32, lane);
      a0 = wmma16(frag_a(H1,          D, ks * 32, lane), b, a0);
      a1 = wmma16(frag_a(H1 + 16 * D, D, ks * 32, lane), b, a1);
    }
    int n = n0 + (lane & 15);
    float b = b2[n];
#pragma unroll
    for (int r = 0; r < 8; ++r) {
      int m = r + ((lane >> 4) << 3);
      float v0 = a0[r] + b, v1 = a1[r] + b;
      H2[m * 64 + n]        = (_Float16)(v0 > 0.0f ? v0 : 0.0f);
      H2[(m + 16) * 64 + n] = (_Float16)(v1 > 0.0f ? v1 : 0.0f);
    }
  }
  __syncthreads();

  if (wave < 2) { // layer 3: 16x16 per wave (col 15 padded)
    const _Float16* src = H2 + wave * 16 * 64;
    v8f acc = {};
#pragma unroll
    for (int ks = 0; ks < 2; ++ks)
      acc = wmma16(frag_a(src, 64, ks * 32, lane), frag_b(w3t, 64, 0, ks * 32, lane), acc);
    int n = lane & 15;
    float b = (n < 15) ? b3[n] : 0.0f;
#pragma unroll
    for (int r = 0; r < 8; ++r) {
      int m = r + ((lane >> 4) << 3) + wave * 16;
      LG[m * 16 + n] = acc[r] + b;
    }
  }
  __syncthreads();

  if (tid < 32) {
    int node = tile + tid;
    if (node < n_nodes) {
      float mx = -3.0e38f;
#pragma unroll
      for (int j = 0; j < 15; ++j) mx = fmaxf(mx, LG[tid * 16 + j]);
      float e[15], s = 0.0f;
#pragma unroll
      for (int j = 0; j < 15; ++j) { e[j] = __expf(LG[tid * 16 + j] - mx); s += e[j]; }
      float inv = 1.0f / s;
#pragma unroll
      for (int j = 0; j < 15; ++j) out[(size_t)node * 15 + j] = e[j] * inv;
    }
  }
}

// ---------------------------------------------------------------------------

extern "C" void kernel_launch(void* const* d_in, const int* in_sizes, int n_in,
                              void* d_out, int out_size, void* d_ws, size_t ws_size,
                              hipStream_t stream) {
  (void)in_sizes; (void)n_in; (void)out_size; (void)ws_size;

  const float* feat    = (const float*)d_in[0];
  const int*   src_ip  = (const int*)d_in[1];   // -> ip_state
  const int*   dst_ip  = (const int*)d_in[2];   // -> conn_state, seg ids for msg1
  const int*   src_c   = (const int*)d_in[3];   // -> conn_state
  const int*   dst_c   = (const int*)d_in[4];   // -> ip_state,   seg ids for msg2
  const float* W_m1    = (const float*)d_in[5];
  const float* b_m1    = (const float*)d_in[6];
  const float* W_m2    = (const float*)d_in[7];
  const float* b_m2    = (const float*)d_in[8];
  const float* g_ip_k  = (const float*)d_in[9];
  const float* g_ip_r  = (const float*)d_in[10];
  const float* g_ip_b  = (const float*)d_in[11];
  const float* g_c_k   = (const float*)d_in[12];
  const float* g_c_r   = (const float*)d_in[13];
  const float* g_c_b   = (const float*)d_in[14];
  const float* W_r1    = (const float*)d_in[15];
  const float* b_r1    = (const float*)d_in[16];
  const float* W_r2    = (const float*)d_in[17];
  const float* b_r2    = (const float*)d_in[18];
  const float* W_r3    = (const float*)d_in[19];
  const float* b_r3    = (const float*)d_in[20];

  // ---- carve workspace (256B aligned chunks) ----
  char* w = (char*)d_ws;
  auto carve = [&](size_t bytes) -> void* {
    void* p = (void*)w;
    w += (bytes + 255) & ~(size_t)255;
    return p;
  };
  float* ip_state   = (float*)carve((size_t)N_IP   * D * 4);
  float* conn_state = (float*)carve((size_t)N_CONN * D * 4);
  float* sum_conn   = (float*)carve((size_t)N_CONN * D * 4);  // ip_mean numerator
  float* sum_ip     = (float*)carve((size_t)N_IP   * D * 4);  // conn_mean numerator
  float* cnt_conn   = (float*)carve((size_t)N_CONN * 4);
  float* cnt_ip     = (float*)carve((size_t)N_IP   * 4);
  _Float16* wm1t  = (_Float16*)carve(128 * 256 * 2);
  _Float16* wm2t  = (_Float16*)carve(128 * 256 * 2);
  _Float16* gipkt = (_Float16*)carve(384 * 128 * 2);
  _Float16* giprt = (_Float16*)carve(384 * 128 * 2);
  _Float16* gckt  = (_Float16*)carve(384 * 128 * 2);
  _Float16* gcrt  = (_Float16*)carve(384 * 128 * 2);
  _Float16* wr1t  = (_Float16*)carve(128 * 128 * 2);
  _Float16* wr2t  = (_Float16*)carve(64 * 128 * 2);
  _Float16* wr3t  = (_Float16*)carve(16 * 64 * 2);

  // ---- 1. weight conversion (f32 -> f16, transposed to [N][K]) ----
  auto tgrid = [](int n) { return (n + 255) / 256; };
  k_transpose_f16<<<tgrid(128 * 256), 256, 0, stream>>>(W_m1,   wm1t,  256, 128, 128);
  k_transpose_f16<<<tgrid(128 * 256), 256, 0, stream>>>(W_m2,   wm2t,  256, 128, 128);
  k_transpose_f16<<<tgrid(384 * 128), 256, 0, stream>>>(g_ip_k, gipkt, 128, 384, 384);
  k_transpose_f16<<<tgrid(384 * 128), 256, 0, stream>>>(g_ip_r, giprt, 128, 384, 384);
  k_transpose_f16<<<tgrid(384 * 128), 256, 0, stream>>>(g_c_k,  gckt,  128, 384, 384);
  k_transpose_f16<<<tgrid(384 * 128), 256, 0, stream>>>(g_c_r,  gcrt,  128, 384, 384);
  k_transpose_f16<<<tgrid(128 * 128), 256, 0, stream>>>(W_r1,   wr1t,  128, 128, 128);
  k_transpose_f16<<<tgrid(64 * 128),  256, 0, stream>>>(W_r2,   wr2t,  128, 64,  64);
  k_transpose_f16<<<tgrid(16 * 64),   256, 0, stream>>>(W_r3,   wr3t,  64,  15,  16);

  // ---- 2. init states ----
  k_init_states<<<tgrid(N_CONN * D), 256, 0, stream>>>(feat, ip_state, conn_state);

  // ---- 3. segment counts (edge->seg topology fixed across iterations) ----
  hipMemsetAsync(cnt_conn, 0, (size_t)N_CONN * 4, stream);
  hipMemsetAsync(cnt_ip,   0, (size_t)N_IP   * 4, stream);
  k_count<<<tgrid(NEDGE), 256, 0, stream>>>(dst_ip, cnt_conn, NEDGE);
  k_count<<<tgrid(NEDGE), 256, 0, stream>>>(dst_c,  cnt_ip,   NEDGE);

  // ---- 4. T message-passing rounds ----
  int eblocks = (NEDGE + 63) / 64;
  for (int t = 0; t < T_ITER; ++t) {
    hipMemsetAsync(sum_conn, 0, (size_t)N_CONN * D * 4, stream);
    hipMemsetAsync(sum_ip,   0, (size_t)N_IP   * D * 4, stream);

    // ip -> connection messages, scattered over connections
    k_msg<<<eblocks, 256, 0, stream>>>(ip_state, src_ip, conn_state, dst_ip,
                                       wm1t, b_m1, sum_conn, NEDGE);
    // connection -> ip messages, scattered over IPs
    k_msg<<<eblocks, 256, 0, stream>>>(conn_state, src_c, ip_state, dst_c,
                                       wm2t, b_m2, sum_ip, NEDGE);

    // GRU updates (both consume pre-update states; in-place per node is safe)
    k_gru<<<(N_IP + 15) / 16, 256, 0, stream>>>(sum_ip, cnt_ip, ip_state,
                                                gipkt, giprt, g_ip_b, N_IP);
    k_gru<<<(N_CONN + 15) / 16, 256, 0, stream>>>(sum_conn, cnt_conn, conn_state,
                                                  gckt, gcrt, g_c_b, N_CONN);
  }

  // ---- 5. readout ----
  k_readout<<<(N_CONN + 31) / 32, 256, 0, stream>>>(
      conn_state, wr1t, b_r1, wr2t, b_r2, wr3t, b_r3, (float*)d_out, N_CONN);
}
